// SAGEBlock_11948599018079
// MI455X (gfx1250) — compile-verified, matrix operand order checked by reading, not run on hardware
//
#include <hip/hip_runtime.h>
#include <hip/hip_bf16.h>

#define D_FEAT 512
#define EPS 1e-5f

typedef __bf16 v16bf __attribute__((ext_vector_type(16)));
typedef __bf16 v8bf  __attribute__((ext_vector_type(8)));
typedef float  v8f   __attribute__((ext_vector_type(8)));

// LDS row stride for the A tile: 1024 bf16 payload + 8 bf16 pad = 2064 bytes.
// 2064 B = 516 dwords; 516 % 64 banks = 4 -> row r starts at bank 4r, so the
// 16 rows hit by lanes 0-15 (and the +16B-shifted lanes 16-31) cover all 64
// banks with no conflicts for ds_load_b128.
#define A_STRIDE 1032

static __device__ __forceinline__ unsigned short f2bf(float f) {
    unsigned u = __float_as_uint(f);
    return (unsigned short)((u + 0x7FFFu + ((u >> 16) & 1u)) >> 16);  // RNE
}

static __device__ __forceinline__ v16bf load_frag(const unsigned short* lo16, const unsigned short* hi16) {
    v8bf lo = *reinterpret_cast<const v8bf*>(lo16);
    v8bf hi = *reinterpret_cast<const v8bf*>(hi16);
    return __builtin_shufflevector(lo, hi, 0,1,2,3,4,5,6,7,8,9,10,11,12,13,14,15);
}

// ---------------- K0: zero scratch (sums, counts, stats) ----------------
__global__ void k_zero(float* __restrict__ p, size_t n) {
    size_t i = (size_t)blockIdx.x * blockDim.x + threadIdx.x;
    size_t stride = (size_t)gridDim.x * blockDim.x;
    for (; i < n; i += stride) p[i] = 0.0f;
}

// ---------------- K1: edge scatter-add (one block per edge) ----------------
__global__ __launch_bounds__(256) void k_scatter(const float* __restrict__ x,
                                                 const long long* __restrict__ ei,
                                                 float* __restrict__ sum,
                                                 float* __restrict__ cnt, int E) {
    int e = blockIdx.x;
    if (e >= E) return;
    int src = (int)ei[e];
    int dst = (int)ei[(size_t)E + e];
    const float* xs = x + (size_t)src * D_FEAT;
    float* sd = sum + (size_t)dst * D_FEAT;
    int t = threadIdx.x;
    atomicAdd(&sd[t],        xs[t]);
    atomicAdd(&sd[t + 256],  xs[t + 256]);
    if (t == 0) atomicAdd(&cnt[dst], 1.0f);
}

// ---------------- K2: aggr = sum/max(cnt,1) -> bf16 ; x -> bf16 ----------------
__global__ __launch_bounds__(256) void k_finalize(const float* __restrict__ sum,
                                                  const float* __restrict__ cnt,
                                                  const float* __restrict__ x,
                                                  unsigned short* __restrict__ abf,
                                                  unsigned short* __restrict__ xbf,
                                                  size_t total) {
    size_t i = (size_t)blockIdx.x * 256 + threadIdx.x;
    if (i >= total) return;
    int row = (int)(i >> 9);
    float inv = 1.0f / fmaxf(cnt[row], 1.0f);
    abf[i] = f2bf(sum[i] * inv);
    xbf[i] = f2bf(x[i]);
}

// ---------------- K3: pack concat weights [Wl|Wr] into WMMA B-operand layout ----
// Layout: [kt(32 over K=1024)][ntile(32 over O=512)][lane(32)][16 bf16 (v0..v7 lo/hi)]
__global__ __launch_bounds__(256) void k_pack(const float* __restrict__ Wl,
                                              const float* __restrict__ Wr,
                                              unsigned short* __restrict__ Wp) {
    int t = blockIdx.x * 256 + threadIdx.x;           // 0 .. 32*32*32-1
    if (t >= 32 * 32 * 32) return;
    int lane  = t & 31;
    int o     = ((t >> 5) & 31) * 16 + (lane & 15);   // output channel
    int kt    = t >> 10;
    int khalf = (lane >> 4) * 8;
    unsigned short* dst = Wp + (size_t)t * 16;
#pragma unroll
    for (int m = 0; m < 16; ++m) {
        int j = m >> 1, h = m & 1;
        int kl = ((j < 4) ? (2 * j + h) : (16 + 2 * (j - 4) + h)) + khalf;
        int k  = kt * 32 + kl;                        // 0..1023 over [aggr|x]
        float w = (k < D_FEAT) ? Wl[(size_t)o * D_FEAT + k]
                               : Wr[(size_t)o * D_FEAT + (k - D_FEAT)];
        dst[m] = f2bf(w);
    }
}

// ---------------- K4: WMMA GEMM H = [aggr|x] @ [Wl|Wr]^T + BN partial stats -----
// Block = 8 waves = one 16-row M tile; wave w owns columns [64w, 64w+64).
// A tile (16x1024 bf16, 33 KB) staged once into LDS, shared by all 8 waves.
// All 4 B fragments of a k-step are loaded into distinct registers before the
// 4 WMMAs so the scheduler can overlap matrix issue with outstanding loads.
__global__ __launch_bounds__(256) void k_gemm(const unsigned short* __restrict__ Abf,
                                              const unsigned short* __restrict__ Xbf,
                                              const unsigned short* __restrict__ Wp,
                                              float* __restrict__ H,
                                              float* __restrict__ Ssum,
                                              float* __restrict__ Ssq) {
    __shared__ unsigned short ldsA[16 * A_STRIDE];    // 33,024 B of 320 KB WGP LDS

    const int lane  = threadIdx.x & 31;
    const int wave  = threadIdx.x >> 5;
    const int mtile = blockIdx.x;
    const int l15   = lane & 15;
    const int lhi   = lane >> 4;
    const int khalf = lhi * 8;

    // --- cooperative stage: 2048 16-byte chunks, 8 per thread, coalesced ---
    {
        const size_t rowbase = (size_t)mtile * 16;
        for (int c = threadIdx.x; c < 2048; c += 256) {
            int r  = c >> 7;               // 128 chunks per row
            int k8 = (c & 127) * 8;        // k offset in elements (0..1016)
            const unsigned short* s =
                (k8 < D_FEAT) ? (Abf + (rowbase + r) * D_FEAT + k8)
                              : (Xbf + (rowbase + r) * D_FEAT + (k8 - D_FEAT));
            *reinterpret_cast<uint4*>(&ldsA[r * A_STRIDE + k8]) =
                *reinterpret_cast<const uint4*>(s);
        }
    }
    __syncthreads();

    v8f acc[4] = {{}, {}, {}, {}};
    const unsigned short* arow = &ldsA[l15 * A_STRIDE];

    for (int kt = 0; kt < 32; ++kt) {
        const unsigned short* s = arow + kt * 32 + khalf;
        v16bf a = load_frag(s, s + 16);   // K 0-7/16-23 (lanes<16), 8-15/24-31 (lanes>=16)
        const unsigned short* wb = Wp + (((size_t)kt * 32 + wave * 4) * 32 + lane) * 16;
        // load all 4 B fragments first (distinct regs -> one 8-load clause,
        // staggered s_wait_loadcnt instead of wait-0 before every WMMA)
        v16bf b0 = load_frag(wb,        wb + 8);
        v16bf b1 = load_frag(wb + 512,  wb + 520);
        v16bf b2 = load_frag(wb + 1024, wb + 1032);
        v16bf b3 = load_frag(wb + 1536, wb + 1544);
        acc[0] = __builtin_amdgcn_wmma_f32_16x16x32_bf16(false, a, false, b0, (short)0, acc[0], false, false);
        acc[1] = __builtin_amdgcn_wmma_f32_16x16x32_bf16(false, a, false, b1, (short)0, acc[1], false, false);
        acc[2] = __builtin_amdgcn_wmma_f32_16x16x32_bf16(false, a, false, b2, (short)0, acc[2], false, false);
        acc[3] = __builtin_amdgcn_wmma_f32_16x16x32_bf16(false, a, false, b3, (short)0, acc[3], false, false);
    }

    const int rbase = mtile * 16 + lhi * 8;
#pragma unroll
    for (int t = 0; t < 4; ++t) {
        const int col = wave * 64 + t * 16 + l15;
        float s = 0.0f, q = 0.0f;
#pragma unroll
        for (int vr = 0; vr < 8; ++vr) {
            float v = acc[t][vr];
            H[(size_t)(rbase + vr) * D_FEAT + col] = v;
            s += v;
            q += v * v;
        }
        atomicAdd(&Ssum[col], s);
        atomicAdd(&Ssq[col], q);
    }
}

// ---------------- K5: fold BN into per-channel scale/shift ----------------
__global__ void k_stats(const float* __restrict__ stats, const float* __restrict__ gamma,
                        const float* __restrict__ beta, float* __restrict__ sa,
                        float* __restrict__ sb, float invN) {
    int c = blockIdx.x * blockDim.x + threadIdx.x;
    if (c >= D_FEAT) return;
    float mu  = stats[c] * invN;
    float var = stats[D_FEAT + c] * invN - mu * mu;
    float a   = gamma[c] * rsqrtf(var + EPS);
    sa[c] = a;
    sb[c] = beta[c] - mu * a;
}

// ---------------- K6: BN apply + ReLU (float4) ----------------
__global__ __launch_bounds__(256) void k_bnrelu(const float4* __restrict__ H4,
                                                const float4* __restrict__ A4,
                                                const float4* __restrict__ B4,
                                                float4* __restrict__ out4, size_t n4) {
    size_t i = (size_t)blockIdx.x * 256 + threadIdx.x;
    if (i >= n4) return;
    int c4 = (int)(i & (size_t)(D_FEAT / 4 - 1));   // 512/4 = 128 float4 per row
    float4 h = H4[i], a = A4[c4], b = B4[c4];
    float4 r;
    r.x = fmaxf(fmaf(h.x, a.x, b.x), 0.0f);
    r.y = fmaxf(fmaf(h.y, a.y, b.y), 0.0f);
    r.z = fmaxf(fmaf(h.z, a.z, b.z), 0.0f);
    r.w = fmaxf(fmaf(h.w, a.w, b.w), 0.0f);
    out4[i] = r;
}

extern "C" void kernel_launch(void* const* d_in, const int* in_sizes, int n_in,
                              void* d_out, int out_size, void* d_ws, size_t ws_size,
                              hipStream_t stream) {
    const float*     x     = (const float*)d_in[0];
    const long long* ei    = (const long long*)d_in[1];   // int64 [2, E]
    const float*     Wl    = (const float*)d_in[2];
    // d_in[3] = b_l : bias cancels exactly under batch-norm, skipped
    const float*     Wr    = (const float*)d_in[4];
    const float*     gamma = (const float*)d_in[5];
    const float*     beta  = (const float*)d_in[6];
    float*           out   = (float*)d_out;

    const int N = in_sizes[0] / D_FEAT;   // 100000
    const int E = in_sizes[1] / 2;        // 800000
    const size_t ND = (size_t)N * D_FEAT;

    char* ws = (char*)d_ws;
    size_t off = 0;
    float* aggrsum = (float*)(ws + off); off += ND * sizeof(float);          // also reused as H
    float* cnt     = (float*)(ws + off); off += (size_t)N * sizeof(float);
    float* stats   = (float*)(ws + off); off += 2048 * sizeof(float);        // Ssum|Ssq|a|b
    off = (off + 255) & ~(size_t)255;
    unsigned short* abf = (unsigned short*)(ws + off); off += ND * sizeof(unsigned short);
    unsigned short* xbf = (unsigned short*)(ws + off); off += ND * sizeof(unsigned short);
    unsigned short* Wp  = (unsigned short*)(ws + off); off += (size_t)1024 * 512 * sizeof(unsigned short);
    float* H  = aggrsum;                  // alias: aggrsum fully consumed before H written
    float* sa = stats + 1024;
    float* sb = stats + 1536;

    // K0: zero sums + counts + stats (must be per-call for graph replay determinism)
    size_t zn = ND + (size_t)N + 2048;
    unsigned zgrid = (unsigned)(((zn + 255) / 256 < 1048576) ? (zn + 255) / 256 : 1048576);
    k_zero<<<zgrid, 256, 0, stream>>>(aggrsum, zn);

    // K1: scatter-add neighbor features + degrees
    k_scatter<<<E, 256, 0, stream>>>(x, ei, aggrsum, cnt, E);

    // K2: mean + bf16 staging
    k_finalize<<<(unsigned)((ND + 255) / 256), 256, 0, stream>>>(aggrsum, cnt, x, abf, xbf, ND);

    // K3: pack weights into WMMA B-operand layout
    k_pack<<<(32 * 32 * 32 + 255) / 256, 256, 0, stream>>>(Wl, Wr, Wp);

    // K4: fused WMMA GEMM (LDS-staged A) + BN partial sums
    k_gemm<<<N / 16, 256, 0, stream>>>(abf, xbf, Wp, H, stats, stats + D_FEAT);

    // K5: BN scale/shift
    k_stats<<<2, 256, 0, stream>>>(stats, gamma, beta, sa, sb, 1.0f / (float)N);

    // K6: apply BN + ReLU
    size_t n4 = ND / 4;
    k_bnrelu<<<(unsigned)((n4 + 255) / 256), 256, 0, stream>>>(
        (const float4*)H, (const float4*)sa, (const float4*)sb, (float4*)out, n4);
}
